// NexusNet_60009283059813
// MI455X (gfx1250) — matrix operand order repaired and checked by reading, not compiled, not run on hardware
//
#include <hip/hip_runtime.h>
#include <math.h>

// ---------------------------------------------------------------------------
// NexusNet for MI455X (gfx1250, wave32, WMMA).
// GEMMs: v_wmma_f32_16x16x32_bf16 (bf16 in, f32 accumulate).
// Weights staged in LDS (padded rows: 136 bf16 = 272B -> <=2-way bank conflicts),
// via GLOBAL_LOAD_ASYNC_TO_LDS_B128 (ASYNCcnt-tracked) when available.
// ---------------------------------------------------------------------------

typedef __attribute__((ext_vector_type(16))) __bf16 v16bf;
typedef __attribute__((ext_vector_type(8)))  __bf16 v8bf;
typedef __attribute__((ext_vector_type(8)))  float  v8f;
typedef int b128v __attribute__((vector_size(16)));   // payload type for async b128

#define WMMA_BF16(a, b, c) \
  __builtin_amdgcn_wmma_f32_16x16x32_bf16(false, (a), false, (b), (short)0, (c), false, false)

#define PP   3
#define FF   128
#define NFF  128
#define NPN  200000
#define EE   600000
#define NXS  40000
#define LDW  136   // padded LDS row stride (bf16 elements): 272B = 68 dwords

#define GLOBAL_AS __attribute__((address_space(1)))
#define LDS_AS    __attribute__((address_space(3)))

#if defined(__AMDGCN__) && __has_builtin(__builtin_amdgcn_global_load_async_to_lds_b128)
#define ASYNC_LDS 1
#else
#define ASYNC_LDS 0
#endif

#if ASYNC_LDS
__device__ __forceinline__ void wait_async0() {
#if __has_builtin(__builtin_amdgcn_s_wait_asynccnt)
  __builtin_amdgcn_s_wait_asynccnt(0);
#else
  asm volatile("s_wait_asynccnt 0" ::: "memory");
#endif
}
#endif

// ---------------------------------------------------------------------------
// Stage a [K][128] bf16 weight matrix from global into LDS with padded stride
// LDW. 256 threads cooperate; K*16 16B-chunks, K multiple of 16 -> uniform trip
// count (EXEC stays full). Async path tracked by ASYNCcnt; sync fallback uses
// b128 global loads + ds stores.
// ---------------------------------------------------------------------------
__device__ __forceinline__ void stage_weights(const __bf16* __restrict__ g,
                                              __bf16* __restrict__ l,
                                              int K, int tid) {
  const int chunks = K * 16;  // 16 x 8-element chunks per row
  for (int c = tid; c < chunks; c += 256) {
    int row = c >> 4;
    int col = (c & 15) * 8;
#if ASYNC_LDS
    __builtin_amdgcn_global_load_async_to_lds_b128(
        (GLOBAL_AS b128v*)(g + (size_t)row * 128 + col),
        (LDS_AS b128v*)(l + (size_t)row * LDW + col),
        0, 0);
#else
    *(v8bf*)(l + (size_t)row * LDW + col) = *(const v8bf*)(g + (size_t)row * 128 + col);
#endif
  }
#if ASYNC_LDS
  wait_async0();
#endif
}

// ---------------------------------------------------------------------------
// Fragment loaders
// ---------------------------------------------------------------------------

// A fragment (16 rows x 32 K) from row-major f32, converted to bf16.
// ISA 16-bit A 16x32 layout: lane&15 = M row; lanes<16 hold K {k0..k0+7,
// k0+16..k0+23}; lanes>=16 the same shifted by 8.
__device__ __forceinline__ v16bf load_a_f32(const float* __restrict__ src, int ld,
                                            int m0, int k0, int lane, int maxRow) {
  int row = m0 + (lane & 15);
  if (row >= maxRow) row = maxRow - 1;          // clamp: keep EXEC all-ones for WMMA
  int kb = k0 + ((lane & 16) ? 8 : 0);
  const float* p = src + (size_t)row * ld + kb;
  const float4 f0 = *(const float4*)(p + 0);
  const float4 f1 = *(const float4*)(p + 4);
  const float4 f2 = *(const float4*)(p + 16);
  const float4 f3 = *(const float4*)(p + 20);
  v16bf a;
  a[0]  = (__bf16)f0.x; a[1]  = (__bf16)f0.y; a[2]  = (__bf16)f0.z; a[3]  = (__bf16)f0.w;
  a[4]  = (__bf16)f1.x; a[5]  = (__bf16)f1.y; a[6]  = (__bf16)f1.z; a[7]  = (__bf16)f1.w;
  a[8]  = (__bf16)f2.x; a[9]  = (__bf16)f2.y; a[10] = (__bf16)f2.z; a[11] = (__bf16)f2.w;
  a[12] = (__bf16)f3.x; a[13] = (__bf16)f3.y; a[14] = (__bf16)f3.z; a[15] = (__bf16)f3.w;
  return a;
}

// A fragment from a bf16 LDS tile (row-major, stride LDW). Rows land on
// distinct bank groups (4-bank shift/row) -> conflict-free 16B ds loads.
__device__ __forceinline__ v16bf load_a_lds(const __bf16* __restrict__ s,
                                            int m0, int k0, int lane) {
  int row = m0 + (lane & 15);
  int kb  = k0 + ((lane & 16) ? 8 : 0);
  const v8bf lo = *(const v8bf*)(s + row * LDW + kb);
  const v8bf hi = *(const v8bf*)(s + row * LDW + kb + 16);
  v16bf a;
#pragma unroll
  for (int i = 0; i < 8; ++i) { a[i] = lo[i]; a[8 + i] = hi[i]; }
  return a;
}

// B fragment (32 K x 16 N) from LDS-staged weights (stride LDW).
// ISA B layout: lane&15 = N column; lanes<16 hold K {k0..k0+7,
// k0+16..k0+23}; lanes>=16 the same shifted by 8.
__device__ __forceinline__ v16bf load_b_lds(const __bf16* __restrict__ w,
                                            int k0, int n0, int lane) {
  const __bf16* p = w + (size_t)(k0 + ((lane & 16) ? 8 : 0)) * LDW + n0 + (lane & 15);
  v16bf b;
#pragma unroll
  for (int i = 0; i < 8; ++i) { b[i] = p[i * LDW]; b[8 + i] = p[(16 + i) * LDW]; }
  return b;
}

// ---------------------------------------------------------------------------
// Kernel 0: f32 -> bf16 weight conversion
// ---------------------------------------------------------------------------
__global__ void cvt_bf16_kernel(const float* __restrict__ src,
                                __bf16* __restrict__ dst, int n) {
  int i = blockIdx.x * blockDim.x + threadIdx.x;
  if (i < n) dst[i] = (__bf16)src[i];
}

// ---------------------------------------------------------------------------
// Kernel 1: nexus_up scatter-sum. One wave per (plane, edge).
// up stored interleaved: up[nx][p*FF + f] => nexus GEMM A rows contiguous.
// ---------------------------------------------------------------------------
__global__ __launch_bounds__(256) void scatter_up_kernel(
    const float* __restrict__ x, const int* __restrict__ esrc,
    const int* __restrict__ edst, float* __restrict__ up) {
  const int lane = threadIdx.x & 31;
  long long wid = (long long)blockIdx.x * 8 + (threadIdx.x >> 5);
  if (wid >= (long long)PP * EE) return;
  int p = (int)(wid / EE);
  int e = (int)(wid % EE);
  int s = esrc[(size_t)p * EE + e];
  int d = edst[(size_t)p * EE + e];
  const float4 v = *(const float4*)(x + ((size_t)p * NPN + s) * FF + lane * 4);
  float* dst = up + (size_t)d * (PP * FF) + p * FF + lane * 4;
  atomicAdd(dst + 0, v.x);
  atomicAdd(dst + 1, v.y);
  atomicAdd(dst + 2, v.z);
  atomicAdd(dst + 3, v.w);
}

// ---------------------------------------------------------------------------
// Kernel 2: fused nexus MLP: n = tanh(tanh(up@W1+b1)@W2+b2).
// 8 waves; block computes 128 rows. W1/W2 staged in LDS; hidden tile reuses
// W1's LDS region after a barrier (W1 dead once GEMM1 finishes).
// LDS: 512*136*2 = 139,264 B.
// ---------------------------------------------------------------------------
__global__ __launch_bounds__(256) void nexus_mlp_kernel(
    const float* __restrict__ up, const __bf16* __restrict__ w1b,
    const float* __restrict__ b1, const __bf16* __restrict__ w2b,
    const float* __restrict__ b2, float* __restrict__ nout) {
  __shared__ __bf16 sW1[384 * LDW];   // GEMM1 weights; first 128 rows reused as sT
  __shared__ __bf16 sW2[128 * LDW];   // GEMM2 weights
  const int tid  = threadIdx.x;
  const int lane = tid & 31;
  const int wave = tid >> 5;
  const int mBlk = blockIdx.x * 128;
  const int mW   = wave * 16;
  const int nlo  = lane & 15;
  const int mhi  = (lane & 16) ? 8 : 0;

  stage_weights(w1b, sW1, 384, tid);
  stage_weights(w2b, sW2, 128, tid);
  __syncthreads();

  v8f acc[8];
#pragma unroll
  for (int i = 0; i < 8; ++i)
#pragma unroll
    for (int r = 0; r < 8; ++r) acc[i][r] = 0.0f;

  for (int k0 = 0; k0 < PP * FF; k0 += 32) {
    v16bf a = load_a_f32(up, PP * FF, mBlk + mW, k0, lane, NXS);
#pragma unroll
    for (int nf = 0; nf < 8; ++nf) {
      v16bf b = load_b_lds(sW1, k0, nf * 16, lane);
      acc[nf] = WMMA_BF16(a, b, acc[nf]);
    }
  }
  __syncthreads();                 // everyone done reading sW1 -> reuse as sT
  __bf16* sT = sW1;

  // C/D layout: element r of v8f -> (M = mW + mhi + r, N = nf*16 + nlo)
#pragma unroll
  for (int nf = 0; nf < 8; ++nf) {
    float bias = b1[nf * 16 + nlo];
#pragma unroll
    for (int r = 0; r < 8; ++r) {
      int row = mW + mhi + r;
      sT[row * LDW + nf * 16 + nlo] = (__bf16)tanhf(acc[nf][r] + bias);
    }
  }
  __syncthreads();

  v8f acc2[8];
#pragma unroll
  for (int i = 0; i < 8; ++i)
#pragma unroll
    for (int r = 0; r < 8; ++r) acc2[i][r] = 0.0f;

  for (int k0 = 0; k0 < NFF; k0 += 32) {
    v16bf a = load_a_lds(sT, mW, k0, lane);
#pragma unroll
    for (int nf = 0; nf < 8; ++nf) {
      v16bf b = load_b_lds(sW2, k0, nf * 16, lane);
      acc2[nf] = WMMA_BF16(a, b, acc2[nf]);
    }
  }

#pragma unroll
  for (int nf = 0; nf < 8; ++nf) {
    float bias = b2[nf * 16 + nlo];
#pragma unroll
    for (int r = 0; r < 8; ++r) {
      int grow = mBlk + mW + mhi + r;
      if (grow < NXS)
        nout[(size_t)grow * NFF + nf * 16 + nlo] = tanhf(acc2[nf][r] + bias);
    }
  }
}

// ---------------------------------------------------------------------------
// Kernel 3 (per plane): edge gate + scatter of gate*n_j and edge counts.
// One wave per edge; 256-dim dot reduced across the wave32.
// ---------------------------------------------------------------------------
__global__ __launch_bounds__(256) void edge_gate_kernel(
    const float* __restrict__ xp, const float* __restrict__ nbuf,
    const int* __restrict__ esrc, const int* __restrict__ edst,
    const float* __restrict__ Wep, const float* __restrict__ bep,
    float* __restrict__ aggr, float* __restrict__ cnt) {
  const int lane = threadIdx.x & 31;
  long long wid = (long long)blockIdx.x * 8 + (threadIdx.x >> 5);
  if (wid >= EE) return;
  int s = esrc[wid];
  int d = edst[wid];
  const float4 xv = *(const float4*)(xp + (size_t)s * FF + lane * 4);
  const float4 nv = *(const float4*)(nbuf + (size_t)d * NFF + lane * 4);
  const float4 w0 = *(const float4*)(Wep + lane * 4);
  const float4 w1 = *(const float4*)(Wep + 128 + lane * 4);
  float dot = xv.x * w0.x + xv.y * w0.y + xv.z * w0.z + xv.w * w0.w +
              nv.x * w1.x + nv.y * w1.y + nv.z * w1.z + nv.w * w1.w;
#pragma unroll
  for (int m = 16; m >= 1; m >>= 1) dot += __shfl_xor(dot, m, 32);
  float gate = 1.0f / (1.0f + __expf(-(dot + bep[0])));
  float* ap = aggr + (size_t)s * NFF + lane * 4;
  atomicAdd(ap + 0, gate * nv.x);
  atomicAdd(ap + 1, gate * nv.y);
  atomicAdd(ap + 2, gate * nv.z);
  atomicAdd(ap + 3, gate * nv.w);
  if (lane == 0) atomicAdd(&cnt[s], 1.0f);
}

// ---------------------------------------------------------------------------
// Kernel 4 (per plane): node MLP: out = tanh(tanh([x|aggr/cnt]@Wn1+bn1)@Wn2+bn2)
// Same fused 2-GEMM-through-LDS structure. LDS: 384*136*2 = 104,448 B.
// ---------------------------------------------------------------------------
__global__ __launch_bounds__(256) void node_mlp_kernel(
    const float* __restrict__ xp, const float* __restrict__ aggr,
    const float* __restrict__ cnt, const __bf16* __restrict__ wn1b,
    const float* __restrict__ bn1, const __bf16* __restrict__ wn2b,
    const float* __restrict__ bn2, float* __restrict__ outp) {
  __shared__ __bf16 sWn1[256 * LDW];  // GEMM1 weights; first 128 rows reused as sT
  __shared__ __bf16 sWn2[128 * LDW];  // GEMM2 weights
  const int tid  = threadIdx.x;
  const int lane = tid & 31;
  const int wave = tid >> 5;
  const int mBlk = blockIdx.x * 128;
  const int mW   = wave * 16;
  const int nlo  = lane & 15;
  const int mhi  = (lane & 16) ? 8 : 0;

  stage_weights(wn1b, sWn1, 256, tid);
  stage_weights(wn2b, sWn2, 128, tid);

  // Each lane's A-row is fixed for the whole K loop -> hoist 1/cnt.
  int row = mBlk + mW + (lane & 15);
  int rowc = row < NPN ? row : NPN - 1;
  size_t roff = (size_t)rowc * FF;
  float inv = 1.0f / fmaxf(cnt[rowc], 1.0f);
  __syncthreads();

  v8f acc[8];
#pragma unroll
  for (int i = 0; i < 8; ++i)
#pragma unroll
    for (int r = 0; r < 8; ++r) acc[i][r] = 0.0f;

  for (int k0 = 0; k0 < 2 * FF; k0 += 32) {
    int kb = k0 + ((lane & 16) ? 8 : 0);
    // chunk0: K = kb..kb+7, chunk1: K = kb+16..kb+23 (never straddle the 128 split)
    const float* s0;  float sc0;
    if (kb < FF)      { s0 = xp + roff + kb;          sc0 = 1.0f; }
    else              { s0 = aggr + roff + (kb - FF); sc0 = inv;  }
    int kb2 = kb + 16;
    const float* s1;  float sc1;
    if (kb2 < FF)     { s1 = xp + roff + kb2;          sc1 = 1.0f; }
    else              { s1 = aggr + roff + (kb2 - FF); sc1 = inv;  }
    const float4 f0 = *(const float4*)(s0 + 0);
    const float4 f1 = *(const float4*)(s0 + 4);
    const float4 f2 = *(const float4*)(s1 + 0);
    const float4 f3 = *(const float4*)(s1 + 4);
    v16bf a;
    a[0]  = (__bf16)(f0.x * sc0); a[1]  = (__bf16)(f0.y * sc0);
    a[2]  = (__bf16)(f0.z * sc0); a[3]  = (__bf16)(f0.w * sc0);
    a[4]  = (__bf16)(f1.x * sc0); a[5]  = (__bf16)(f1.y * sc0);
    a[6]  = (__bf16)(f1.z * sc0); a[7]  = (__bf16)(f1.w * sc0);
    a[8]  = (__bf16)(f2.x * sc1); a[9]  = (__bf16)(f2.y * sc1);
    a[10] = (__bf16)(f2.z * sc1); a[11] = (__bf16)(f2.w * sc1);
    a[12] = (__bf16)(f3.x * sc1); a[13] = (__bf16)(f3.y * sc1);
    a[14] = (__bf16)(f3.z * sc1); a[15] = (__bf16)(f3.w * sc1);
#pragma unroll
    for (int nf = 0; nf < 8; ++nf) {
      v16bf b = load_b_lds(sWn1, k0, nf * 16, lane);
      acc[nf] = WMMA_BF16(a, b, acc[nf]);
    }
  }
  __syncthreads();                 // done reading sWn1 -> reuse as sT
  __bf16* sT = sWn1;

#pragma unroll
  for (int nf = 0; nf < 8; ++nf) {
    float bias = bn1[nf * 16 + nlo];
#pragma unroll
    for (int r = 0; r < 8; ++r) {
      int rloc = mW + mhi + r;
      sT[rloc * LDW + nf * 16 + nlo] = (__bf16)tanhf(acc[nf][r] + bias);
    }
  }
  __syncthreads();

  v8f acc2[8];
#pragma unroll
  for (int i = 0; i < 8; ++i)
#pragma unroll
    for (int r = 0; r < 8; ++r) acc2[i][r] = 0.0f;

  for (int k0 = 0; k0 < FF; k0 += 32) {
    v16bf a = load_a_lds(sT, mW, k0, lane);
#pragma unroll
    for (int nf = 0; nf < 8; ++nf) {
      v16bf b = load_b_lds(sWn2, k0, nf * 16, lane);
      acc2[nf] = WMMA_BF16(a, b, acc2[nf]);
    }
  }

#pragma unroll
  for (int nf = 0; nf < 8; ++nf) {
    float bias = bn2[nf * 16 + nlo];
#pragma unroll
    for (int r = 0; r < 8; ++r) {
      int grow = mBlk + mW + mhi + r;
      if (grow < NPN)
        outp[(size_t)grow * FF + nf * 16 + nlo] = tanhf(acc2[nf][r] + bias);
    }
  }
}

// ---------------------------------------------------------------------------
// Launcher
// ---------------------------------------------------------------------------
extern "C" void kernel_launch(void* const* d_in, const int* in_sizes, int n_in,
                              void* d_out, int out_size, void* d_ws, size_t ws_size,
                              hipStream_t stream) {
  (void)in_sizes; (void)n_in; (void)out_size; (void)ws_size;
  const float* x    = (const float*)d_in[0];   // [3,200000,128]
  const int*   esrc = (const int*)d_in[1];     // [3,600000]
  const int*   edst = (const int*)d_in[2];     // [3,600000]
  const float* W1   = (const float*)d_in[3];   // [384,128]
  const float* b1   = (const float*)d_in[4];   // [128]
  const float* W2   = (const float*)d_in[5];   // [128,128]
  const float* b2   = (const float*)d_in[6];   // [128]
  const float* We   = (const float*)d_in[7];   // [3,256,1]
  const float* be   = (const float*)d_in[8];   // [3,1]
  const float* Wn1  = (const float*)d_in[9];   // [3,256,128]
  const float* bn1  = (const float*)d_in[10];  // [3,128]
  const float* Wn2  = (const float*)d_in[11];  // [3,128,128]
  const float* bn2  = (const float*)d_in[12];  // [3,128]
  float* out = (float*)d_out;                  // [3,200000,128]

  // Workspace carve-up (256B-aligned by construction).
  char* w = (char*)d_ws;
  float* up   = (float*)w; w += (size_t)NXS * PP * FF * sizeof(float);  // 61.44 MB
  float* nbuf = (float*)w; w += (size_t)NXS * NFF * sizeof(float);      // 20.48 MB
  float* aggr = (float*)w; w += (size_t)NPN * NFF * sizeof(float);      // 102.4 MB (per-plane reuse)
  float* cnt  = (float*)w; w += (size_t)NPN * sizeof(float);            // 0.8 MB
  __bf16* w1b  = (__bf16*)w; w += (size_t)384 * 128 * sizeof(__bf16);
  __bf16* w2b  = (__bf16*)w; w += (size_t)128 * 128 * sizeof(__bf16);
  __bf16* wn1b = (__bf16*)w; w += (size_t)PP * 256 * 128 * sizeof(__bf16);
  __bf16* wn2b = (__bf16*)w; w += (size_t)PP * 128 * 128 * sizeof(__bf16);

  // Weight conversion (tiny).
  cvt_bf16_kernel<<<(384 * 128 + 255) / 256, 256, 0, stream>>>(W1, w1b, 384 * 128);
  cvt_bf16_kernel<<<(128 * 128 + 255) / 256, 256, 0, stream>>>(W2, w2b, 128 * 128);
  cvt_bf16_kernel<<<(PP * 256 * 128 + 255) / 256, 256, 0, stream>>>(Wn1, wn1b, PP * 256 * 128);
  cvt_bf16_kernel<<<(PP * 128 * 128 + 255) / 256, 256, 0, stream>>>(Wn2, wn2b, PP * 128 * 128);

  // Stage 1: nexus_up scatter-sum.
  (void)hipMemsetAsync(up, 0, (size_t)NXS * PP * FF * sizeof(float), stream);
  {
    long long waves = (long long)PP * EE;
    int blocks = (int)((waves + 7) / 8);
    scatter_up_kernel<<<blocks, 256, 0, stream>>>(x, esrc, edst, up);
  }

  // Stage 2: fused nexus MLP.
  nexus_mlp_kernel<<<(NXS + 127) / 128, 256, 0, stream>>>(up, w1b, b1, w2b, b2, nbuf);

  // Stage 3: per-plane gate/scatter-mean + node MLP (aggr/cnt reused).
  for (int p = 0; p < PP; ++p) {
    (void)hipMemsetAsync(aggr, 0, (size_t)NPN * NFF * sizeof(float), stream);
    (void)hipMemsetAsync(cnt, 0, (size_t)NPN * sizeof(float), stream);
    edge_gate_kernel<<<(EE + 7) / 8, 256, 0, stream>>>(
        x + (size_t)p * NPN * FF, nbuf, esrc + (size_t)p * EE, edst + (size_t)p * EE,
        We + (size_t)p * 256, be + p, aggr, cnt);
    node_mlp_kernel<<<(NPN + 127) / 128, 256, 0, stream>>>(
        x + (size_t)p * NPN * FF, aggr, cnt, wn1b + (size_t)p * 256 * 128,
        bn1 + (size_t)p * 128, wn2b + (size_t)p * 128 * 128, bn2 + (size_t)p * 128,
        out + (size_t)p * NPN * FF);
  }
}
